// SecretRQVAE_51513837748676
// MI455X (gfx1250) — matrix operand order repaired and checked by reading, not compiled, not run on hardware
//
#include <hip/hip_runtime.h>
#include <hip/hip_bf16.h>
#include <math.h>

typedef __attribute__((ext_vector_type(16))) _Float16 v16h;
typedef __attribute__((ext_vector_type(8)))  _Float16 v8h;
typedef __attribute__((ext_vector_type(8)))  float    v8f;

#define BATCH 8
#define DEMB  256
#define NCODE 1024
#define NQ    4
#define NTOK  (BATCH * 64 * 64)   // 32768 latent tokens

#define SHUF16(lo, hi) __builtin_shufflevector((lo), (hi), \
    0, 1, 2, 3, 4, 5, 6, 7, 8, 9, 10, 11, 12, 13, 14, 15)

// ---------------------------------------------------------------------------
// Generic implicit-GEMM convolution, NCHW activations, OIHW weights.
// Workgroup: 256 threads = 8 waves, tile 64 (Cout) x 64 (output pixels).
// Each wave computes TWO 16x16 f32 tiles (M and M+16) sharing one B fragment:
// 2 WMMAs per 3 b128 LDS fragment loads, halving redundant global reads
// versus a 32-row M tile.
// ---------------------------------------------------------------------------
struct ConvP {
  int Cin, Cout, Hin, Win, Hout, Wout, KH, KW, stride, pad, act; // act 0=none 1=relu
};

__global__ void __launch_bounds__(256) conv_wmma(
    const float* __restrict__ X, const float* __restrict__ W,
    const float* __restrict__ Bv, float* __restrict__ Y, ConvP p)
{
  const int K     = p.Cin * p.KH * p.KW;
  const int Ks    = (K + 31) >> 5;              // K padded to multiple of 32
  const int Ncols = BATCH * p.Hout * p.Wout;
  const int n0    = blockIdx.x * 64;
  const int m0    = blockIdx.y * 64;
  const int tid   = threadIdx.x;
  const int lane  = tid & 31, wv = tid >> 5;
  const int wm    = wv >> 2,  wn = wv & 3;      // 2 x 4 wave grid

  __shared__ __attribute__((aligned(16))) _Float16 As[64][40];   // [m][k], 80B row
  __shared__ __attribute__((aligned(16))) _Float16 BsT[64][40];  // [n][k], 80B row

  v8f acc0 = {}, acc1 = {};

  for (int ks = 0; ks < Ks; ++ks) {
    const int kb = ks << 5;
    __syncthreads();
    { // A tile: weights, K-contiguous in OIHW (64 rows x 32 k)
      int row = tid >> 2, c0 = (tid & 3) << 3;
      int m = m0 + row;
      #pragma unroll
      for (int j = 0; j < 8; ++j) {
        int k = kb + c0 + j;
        float v = 0.f;
        if (m < p.Cout && k < K) v = W[m * K + k];
        As[row][c0 + j] = (_Float16)v;
      }
    }
    { // B tile: implicit im2col gather from NCHW input, stored [n][k]
      int krow = tid >> 3, c0 = (tid & 7) << 3;
      int k = kb + krow;
      int ci = 0, kh = 0, kw = 0;
      bool kok = (k < K);
      if (kok) {
        int kk = p.KH * p.KW;
        ci = k / kk; int r = k - ci * kk; kh = r / p.KW; kw = r - kh * p.KW;
      }
      #pragma unroll
      for (int j = 0; j < 8; ++j) {
        int n = n0 + c0 + j;
        float v = 0.f;
        if (kok && n < Ncols) {
          int b  = n / (p.Hout * p.Wout);
          int hw = n - b * p.Hout * p.Wout;
          int oh = hw / p.Wout, ow = hw - oh * p.Wout;
          int ih = oh * p.stride - p.pad + kh;
          int iw = ow * p.stride - p.pad + kw;
          if (ih >= 0 && ih < p.Hin && iw >= 0 && iw < p.Win)
            v = X[((b * p.Cin + ci) * p.Hin + ih) * p.Win + iw];
        }
        BsT[c0 + j][krow] = (_Float16)v;
      }
    }
    __syncthreads();
    // Fragment loads: 2x ds_load_b128 each, per CDNA5 ISA VGPR layouts (wave32)
    int mr = (wm << 5) + (lane & 15);
    const v8h* ar0 = (const v8h*)(&As[mr][0]);
    const v8h* ar1 = (const v8h*)(&As[mr + 16][0]);
    const v8h* br  = (const v8h*)(&BsT[(wn << 4) + (lane & 15)][0]);
    int ja = (lane < 16) ? 0 : 1;      // A halves [8ja..], [8ja+16..]
    int jb = (lane < 16) ? 0 : 1;      // B halves [8jb..], [8jb+16..]
    v16h af0 = SHUF16(ar0[ja], ar0[ja + 2]);
    v16h af1 = SHUF16(ar1[ja], ar1[ja + 2]);
    v16h bf  = SHUF16(br[jb], br[jb + 2]);
    acc0 = __builtin_amdgcn_wmma_f32_16x16x32_f16(false, af0, false, bf,
                                                  (short)0, acc0, false, false);
    acc1 = __builtin_amdgcn_wmma_f32_16x16x32_f16(false, af1, false, bf,
                                                  (short)0, acc1, false, false);
  }

  // Epilogue: bias + activation, store NCHW
  int n = n0 + (wn << 4) + (lane & 15);
  if (n < Ncols) {
    int b  = n / (p.Hout * p.Wout);
    int hw = n - b * p.Hout * p.Wout;
    int oh = hw / p.Wout, ow = hw - oh * p.Wout;
    int mb = m0 + (wm << 5) + ((lane < 16) ? 0 : 8);
    #pragma unroll
    for (int r = 0; r < 8; ++r) {
      int m = mb + r;
      if (m < p.Cout) {
        float v = acc0[r] + Bv[m];
        if (p.act == 1) v = fmaxf(v, 0.f);
        Y[((b * p.Cout + m) * p.Hout + oh) * p.Wout + ow] = v;
      }
      int m2 = mb + 16 + r;
      if (m2 < p.Cout) {
        float v = acc1[r] + Bv[m2];
        if (p.act == 1) v = fmaxf(v, 0.f);
        Y[((b * p.Cout + m2) * p.Hout + oh) * p.Wout + ow] = v;
      }
    }
  }
}

// ---------------------------------------------------------------------------
// ConvTranspose2d k4 s2 p1 (exact 2x upsample), IOHW weights.
// Decomposed into 4 output-parity classes (grid.z); each class is an
// implicit GEMM with K = Cin*4 (the 4 valid taps per output pixel).
// Same dual-accumulator 64x64 tiling as conv_wmma. act: 1=relu, 2=tanh
// ---------------------------------------------------------------------------
struct ConvTP { int Cin, Cout, Hin, Win, Hout, Wout, act; };

__global__ void __launch_bounds__(256) convt_wmma(
    const float* __restrict__ X, const float* __restrict__ Wt,
    const float* __restrict__ Bv, float* __restrict__ Y, ConvTP p)
{
  const int poh = (int)blockIdx.z >> 1, pw = (int)blockIdx.z & 1;
  const int kh0 = poh ^ 1, kw0 = pw ^ 1;          // kh parity == (oh+1)&1
  const int Hq = p.Hout >> 1, Wq = p.Wout >> 1;
  const int K  = p.Cin * 4;                       // multiple of 32 for all layers
  const int Ks = K >> 5;
  const int Ncols = BATCH * Hq * Wq;
  const int n0 = blockIdx.x * 64;
  const int m0 = blockIdx.y * 64;
  const int tid  = threadIdx.x;
  const int lane = tid & 31, wv = tid >> 5;
  const int wm = wv >> 2, wn = wv & 3;

  __shared__ __attribute__((aligned(16))) _Float16 As[64][40];
  __shared__ __attribute__((aligned(16))) _Float16 BsT[64][40];

  v8f acc0 = {}, acc1 = {};

  for (int ks = 0; ks < Ks; ++ks) {
    const int kb = ks << 5;
    __syncthreads();
    { // A tile: gather weights for this parity class; IOHW layout
      int row = tid >> 2, c0 = (tid & 3) << 3;
      int m = m0 + row;
      #pragma unroll
      for (int j = 0; j < 8; ++j) {
        int k = kb + c0 + j;
        float v = 0.f;
        if (m < p.Cout && k < K) {
          int ci = k >> 2, t = k & 3;
          int kh = kh0 + ((t >> 1) << 1);
          int kw = kw0 + ((t & 1) << 1);
          v = Wt[((ci * p.Cout + m) * 4 + kh) * 4 + kw];
        }
        As[row][c0 + j] = (_Float16)v;
      }
    }
    { // B tile: gather input taps for quadrant pixels, stored [n][k]
      int krow = tid >> 3, c0 = (tid & 7) << 3;
      int k = kb + krow;
      int ci = k >> 2, t = k & 3;
      int kh = kh0 + ((t >> 1) << 1);
      int kw = kw0 + ((t & 1) << 1);
      #pragma unroll
      for (int j = 0; j < 8; ++j) {
        int n = n0 + c0 + j;
        float v = 0.f;
        if (n < Ncols) {
          int b  = n / (Hq * Wq);
          int hw = n - b * Hq * Wq;
          int oh = ((hw / Wq) << 1) + poh;
          int ow = ((hw - (hw / Wq) * Wq) << 1) + pw;
          int ih2 = oh + 1 - kh, iw2 = ow + 1 - kw;   // even by construction
          int ih = ih2 >> 1, iw = iw2 >> 1;
          if (ih2 >= 0 && ih < p.Hin && iw2 >= 0 && iw < p.Win)
            v = X[((b * p.Cin + ci) * p.Hin + ih) * p.Win + iw];
        }
        BsT[c0 + j][krow] = (_Float16)v;
      }
    }
    __syncthreads();
    int mr = (wm << 5) + (lane & 15);
    const v8h* ar0 = (const v8h*)(&As[mr][0]);
    const v8h* ar1 = (const v8h*)(&As[mr + 16][0]);
    const v8h* br  = (const v8h*)(&BsT[(wn << 4) + (lane & 15)][0]);
    int ja = (lane < 16) ? 0 : 1;
    int jb = (lane < 16) ? 0 : 1;
    v16h af0 = SHUF16(ar0[ja], ar0[ja + 2]);
    v16h af1 = SHUF16(ar1[ja], ar1[ja + 2]);
    v16h bf  = SHUF16(br[jb], br[jb + 2]);
    acc0 = __builtin_amdgcn_wmma_f32_16x16x32_f16(false, af0, false, bf,
                                                  (short)0, acc0, false, false);
    acc1 = __builtin_amdgcn_wmma_f32_16x16x32_f16(false, af1, false, bf,
                                                  (short)0, acc1, false, false);
  }

  int n = n0 + (wn << 4) + (lane & 15);
  if (n < Ncols) {
    int b  = n / (Hq * Wq);
    int hw = n - b * Hq * Wq;
    int oh = ((hw / Wq) << 1) + poh;
    int ow = ((hw - (hw / Wq) * Wq) << 1) + pw;
    int mb = m0 + (wm << 5) + ((lane < 16) ? 0 : 8);
    #pragma unroll
    for (int r = 0; r < 8; ++r) {
      int m = mb + r;
      if (m < p.Cout) {
        float v = acc0[r] + Bv[m];
        if (p.act == 1) v = fmaxf(v, 0.f);
        else if (p.act == 2) v = tanhf(v);
        Y[((b * p.Cout + m) * p.Hout + oh) * p.Wout + ow] = v;
      }
      int m2 = mb + 16 + r;
      if (m2 < p.Cout) {
        float v = acc1[r] + Bv[m2];
        if (p.act == 1) v = fmaxf(v, 0.f);
        else if (p.act == 2) v = tanhf(v);
        Y[((b * p.Cout + m2) * p.Hout + oh) * p.Wout + ow] = v;
      }
    }
  }
}

// ---------------------------------------------------------------------------
// RQ stage 1: distances + argmin over 1024 codes for 16 tokens per WG.
// d2 = ||c||^2 - 2 r.c (token norm constant -> irrelevant for argmin).
// 8 waves sweep 128 codes per pass, 8 passes; K=256 in 8 WMMA steps.
// ---------------------------------------------------------------------------
__global__ void __launch_bounds__(256) rq_argmin(
    const float* __restrict__ Res, const float* __restrict__ CB,
    const float* __restrict__ C2, int* __restrict__ idxI,
    float* __restrict__ idxF, int q)
{
  __shared__ __attribute__((aligned(16))) _Float16 As[16][DEMB + 8]; // [m][d], 528B row
  __shared__ __attribute__((aligned(16))) _Float16 BsT[128][40];     // [code][k]
  __shared__ float cv[8][16];
  __shared__ int   cidx[8][16];
  const int tid = threadIdx.x, lane = tid & 31, wv = tid >> 5;
  const int t0 = blockIdx.x << 4;

  for (int i = tid; i < 16 * DEMB; i += 256) {
    int m = i >> 8, d = i & 255;
    As[m][d] = (_Float16)Res[(t0 + m) * DEMB + d];
  }

  float mv[8]; int mi[8];
  #pragma unroll
  for (int r = 0; r < 8; ++r) { mv[r] = 3.0e38f; mi[r] = 0; }

  const v8h* ar = (const v8h*)(&As[lane & 15][0]);
  const int ja = (lane < 16) ? 0 : 1;
  const int jb = (lane < 16) ? 0 : 1;

  for (int pass = 0; pass < 8; ++pass) {
    v8f acc = {};
    for (int ks = 0; ks < 8; ++ks) {
      __syncthreads();
      for (int i = tid; i < 32 * 128; i += 256) {
        int kr = i & 31, col = i >> 5;   // consecutive tid -> consecutive dim (coalesced)
        BsT[col][kr] = (_Float16)CB[((pass << 7) + col) * DEMB + (ks << 5) + kr];
      }
      __syncthreads();
      const v8h* br = (const v8h*)(&BsT[(wv << 4) + (lane & 15)][0]);
      v16h af = SHUF16(ar[(ks << 2) + ja], ar[(ks << 2) + ja + 2]);
      v16h bf = SHUF16(br[jb], br[jb + 2]);
      acc = __builtin_amdgcn_wmma_f32_16x16x32_f16(false, af, false, bf,
                                                   (short)0, acc, false, false);
    }
    int ncode = (pass << 7) + (wv << 4) + (lane & 15);
    float cn = C2[ncode];
    #pragma unroll
    for (int r = 0; r < 8; ++r) {
      float s = cn - 2.0f * acc[r];
      if (s < mv[r]) { mv[r] = s; mi[r] = ncode; }
    }
  }
  // argmin across the 16 lanes of each half-wave (same token, different codes)
  #pragma unroll
  for (int msk = 1; msk <= 8; msk <<= 1) {
    #pragma unroll
    for (int r = 0; r < 8; ++r) {
      float ov = __shfl_xor(mv[r], msk, 32);
      int   oi = __shfl_xor(mi[r], msk, 32);
      if (ov < mv[r] || (ov == mv[r] && oi < mi[r])) { mv[r] = ov; mi[r] = oi; }
    }
  }
  if ((lane & 15) == 0) {
    int off = (lane < 16) ? 0 : 8;
    #pragma unroll
    for (int r = 0; r < 8; ++r) { cv[wv][off + r] = mv[r]; cidx[wv][off + r] = mi[r]; }
  }
  __syncthreads();
  if (tid < 16) {  // cross-wave reduce, write indices
    float bv = cv[0][tid]; int bi = cidx[0][tid];
    #pragma unroll
    for (int w = 1; w < 8; ++w) {
      float v = cv[w][tid]; int ii = cidx[w][tid];
      if (v < bv || (v == bv && ii < bi)) { bv = v; bi = ii; }
    }
    int tok = t0 + tid;
    idxI[tok] = bi;
    int b = tok >> 12, nn = tok & 4095;       // N = 4096 tokens per batch
    idxF[((b * NQ + q) << 12) + nn] = (float)bi;
  }
}

// RQ stage 2: gather codes, STE forward update, per-token loss partials.
__global__ void __launch_bounds__(256) rq_update(
    const float* __restrict__ CB, const int* __restrict__ idxI,
    float* __restrict__ Res, float* __restrict__ Qnt,
    float* __restrict__ lpart)
{
  __shared__ float red[256];
  int t = blockIdx.x, d = threadIdx.x;
  int id = idxI[t];
  float qv = CB[id * DEMB + d];
  float rv = Res[t * DEMB + d];
  float df = qv - rv;
  Qnt[t * DEMB + d] += qv;        // quant + residual + sg(q - residual) == quant + q
  Res[t * DEMB + d] = rv - qv;
  red[d] = df * df;
  __syncthreads();
  for (int s = 128; s > 0; s >>= 1) {
    if (d < s) red[d] += red[d + s];
    __syncthreads();
  }
  if (d == 0) lpart[t] = red[0];
}

// Deterministic fixed-order loss reduction (no float atomics).
__global__ void __launch_bounds__(256) loss_reduce(
    const float* __restrict__ lpart, float* __restrict__ out)
{
  __shared__ float red[256];
  float s = 0.f;
  for (int i = threadIdx.x; i < NTOK; i += 256) s += lpart[i];
  red[threadIdx.x] = s;
  __syncthreads();
  for (int st = 128; st > 0; st >>= 1) {
    if (threadIdx.x < st) red[threadIdx.x] += red[threadIdx.x + st];
    __syncthreads();
  }
  if (threadIdx.x == 0) *out = red[0] * (1.0f / (float)(NTOK * DEMB));
}

// Codebook squared norms, all Q*K codes.
__global__ void __launch_bounds__(256) cb_norm_k(
    const float* __restrict__ CB, float* __restrict__ C2)
{
  int c = blockIdx.x * 256 + threadIdx.x;
  if (c >= NQ * NCODE) return;
  const float* p = CB + (size_t)c * DEMB;
  float s = 0.f;
  for (int d = 0; d < DEMB; ++d) s += p[d] * p[d];
  C2[c] = s;
}

// z NCHW [8,256,64,64] -> residual [B,N,D]; zero quant.
__global__ void __launch_bounds__(256) z_to_res(
    const float* __restrict__ z, float* __restrict__ res, float* __restrict__ qnt)
{
  int i = blockIdx.x * 256 + threadIdx.x;
  if (i >= BATCH * DEMB * 64 * 64) return;
  int w = i & 63, h = (i >> 6) & 63, d = (i >> 12) & 255, b = i >> 20;
  int o = (((b << 12) + (h << 6) + w) << 8) + d;   // (b*4096 + h*64 + w)*256 + d
  res[o] = z[i];
  qnt[o] = 0.f;
}

// quant [B,N,D] -> NCHW, written to both decoder input and d_out.
__global__ void __launch_bounds__(256) quant_to_nchw(
    const float* __restrict__ qnt, float* __restrict__ qnchw, float* __restrict__ qout)
{
  int i = blockIdx.x * 256 + threadIdx.x;
  if (i >= BATCH * DEMB * 64 * 64) return;
  int w = i & 63, h = (i >> 6) & 63, d = (i >> 12) & 255, b = i >> 20;
  float v = qnt[(((b << 12) + (h << 6) + w) << 8) + d];
  qnchw[i] = v;
  qout[i] = v;
}

// ---------------------------------------------------------------------------
extern "C" void kernel_launch(void* const* d_in, const int* in_sizes, int n_in,
                              void* d_out, int out_size, void* d_ws, size_t ws_size,
                              hipStream_t stream) {
  // Inputs in setup_inputs() insertion order:
  // [0] x; [1..8] enc (w,b)x4; [9] codebooks [4,1024,256]; [10..17] dec (w,b)x4
  const float* x   = (const float*)d_in[0];
  const float* ew0 = (const float*)d_in[1];  const float* eb0 = (const float*)d_in[2];
  const float* ew1 = (const float*)d_in[3];  const float* eb1 = (const float*)d_in[4];
  const float* ew2 = (const float*)d_in[5];  const float* eb2 = (const float*)d_in[6];
  const float* ew3 = (const float*)d_in[7];  const float* eb3 = (const float*)d_in[8];
  const float* CB  = (const float*)d_in[9];
  const float* dw0 = (const float*)d_in[10]; const float* db0 = (const float*)d_in[11];
  const float* dw1 = (const float*)d_in[12]; const float* db1 = (const float*)d_in[13];
  const float* dw2 = (const float*)d_in[14]; const float* db2 = (const float*)d_in[15];
  const float* dw3 = (const float*)d_in[16]; const float* db3 = (const float*)d_in[17];

  // Workspace layout (floats), with reuse: d2<-a1, dec1<-a2, qnchw<-a3
  float* ws  = (float*)d_ws;
  float* a1  = ws;               // 8*64*128*128  = 8388608
  float* a2  = a1 + 8388608;     // 8*128*64*64   = 4194304
  float* a3  = a2 + 4194304;     // 8*256*64*64   = 8388608
  float* zb  = a3 + 8388608;     // 8388608
  float* res = zb + 8388608;     // 8388608
  float* qnt = res + 8388608;    // 8388608
  float* d3  = qnt + 8388608;    // 8*64*256*256  = 33554432
  float* c2  = d3 + 33554432;    // 4096
  float* lpart = c2 + 4096;      // 32768
  int*   idxI  = (int*)(lpart + NTOK); // 32768 ints

  // Output layout: recon | indices | commit_loss | quantized
  float* out   = (float*)d_out;
  float* recon = out;                          // 8*3*512*512 = 6291456
  float* idxF  = out + 6291456;                // 8*4*64*64   = 131072
  float* lossO = idxF + 131072;                // 4
  float* qout  = lossO + 4;                    // 8*256*64*64 = 8388608

  dim3 blk(256);

  // ---- Encoder ----
  { ConvP p{3, 64, 256, 256, 128, 128, 4, 4, 2, 1, 1};
    dim3 g((BATCH*128*128 + 63) / 64, (64 + 63) / 64);
    conv_wmma<<<g, blk, 0, stream>>>(x, ew0, eb0, a1, p); }
  { ConvP p{64, 128, 128, 128, 64, 64, 4, 4, 2, 1, 1};
    dim3 g((BATCH*64*64 + 63) / 64, (128 + 63) / 64);
    conv_wmma<<<g, blk, 0, stream>>>(a1, ew1, eb1, a2, p); }
  { ConvP p{128, 256, 64, 64, 64, 64, 3, 3, 1, 1, 1};
    dim3 g((BATCH*64*64 + 63) / 64, (256 + 63) / 64);
    conv_wmma<<<g, blk, 0, stream>>>(a2, ew2, eb2, a3, p); }
  { ConvP p{256, 256, 64, 64, 64, 64, 3, 3, 1, 1, 0};
    dim3 g((BATCH*64*64 + 63) / 64, (256 + 63) / 64);
    conv_wmma<<<g, blk, 0, stream>>>(a3, ew3, eb3, zb, p); }

  // ---- Residual quantization ----
  cb_norm_k<<<(NQ*NCODE + 255) / 256, blk, 0, stream>>>(CB, c2);
  z_to_res<<<(BATCH*DEMB*64*64 + 255) / 256, blk, 0, stream>>>(zb, res, qnt);
  for (int q = 0; q < NQ; ++q) {
    const float* cbq = CB + (size_t)q * NCODE * DEMB;
    rq_argmin<<<NTOK / 16, blk, 0, stream>>>(res, cbq, c2 + q * NCODE, idxI, idxF, q);
    rq_update<<<NTOK, blk, 0, stream>>>(cbq, idxI, res, qnt, lpart);
    loss_reduce<<<1, blk, 0, stream>>>(lpart, lossO + q);
  }
  float* qnchw = a3;   // reuse
  quant_to_nchw<<<(BATCH*DEMB*64*64 + 255) / 256, blk, 0, stream>>>(qnt, qnchw, qout);

  // ---- Decoder ----
  float* dec1 = a2;    // reuse
  { ConvP p{256, 128, 64, 64, 64, 64, 3, 3, 1, 1, 1};
    dim3 g((BATCH*64*64 + 63) / 64, (128 + 63) / 64);
    conv_wmma<<<g, blk, 0, stream>>>(qnchw, dw0, db0, dec1, p); }
  float* d2b = a1;     // reuse
  { ConvTP p{128, 64, 64, 64, 128, 128, 1};
    dim3 g((BATCH*64*64 + 63) / 64, 1, 4);
    convt_wmma<<<g, blk, 0, stream>>>(dec1, dw1, db1, d2b, p); }
  { ConvTP p{64, 64, 128, 128, 256, 256, 1};
    dim3 g((BATCH*128*128 + 63) / 64, 1, 4);
    convt_wmma<<<g, blk, 0, stream>>>(d2b, dw2, db2, d3, p); }
  { ConvTP p{64, 3, 256, 256, 512, 512, 2};   // tanh fused
    dim3 g((BATCH*256*256 + 63) / 64, 1, 4);
    convt_wmma<<<g, blk, 0, stream>>>(d3, dw3, db3, recon, p); }
}
